// Transformer_70781061038468
// MI455X (gfx1250) — matrix operand order repaired and checked by reading, not compiled, not run on hardware
//
#include <hip/hip_runtime.h>
#include <hip/hip_bf16.h>
#include <math.h>

// ---------------------------------------------------------------------------
// CDNA5 (gfx1250) bf16 WMMA transformer block.
// D = A x B + C via v_wmma_f32_16x16x32_bf16, wave32, 8-wave workgroups.
// GEMM tile staging uses GLOBAL_LOAD_ASYNC_TO_LDS (ASYNCcnt) when the
// toolchain exposes the builtin; otherwise falls back to load+ds_store.
// ---------------------------------------------------------------------------

typedef __attribute__((ext_vector_type(16))) __bf16 v16bf;
typedef __attribute__((ext_vector_type(8)))  float  v8f;

#if defined(__AMDGCN__) && \
    __has_builtin(__builtin_amdgcn_global_load_async_to_lds_b128) && \
    __has_builtin(__builtin_amdgcn_s_wait_asynccnt)
#define USE_ASYNC_LDS 1
#else
#define USE_ASYNC_LDS 0
#endif

union Frag {
    v16bf v;
    uint4 q[2];
    unsigned short s[16];
};

__device__ __forceinline__ unsigned short f2bf(float f) {
    union { float f; unsigned int u; } c; c.f = f;
    unsigned int u = c.u;
    u += 0x7FFFu + ((u >> 16) & 1u);   // round-to-nearest-even
    return (unsigned short)(u >> 16);
}

__device__ __forceinline__ float gelu_exact(float x) {
    return 0.5f * x * (1.0f + erff(x * 0.70710678118654752f));
}

#if USE_ASYNC_LDS
// Builtin signature (from hipcc diagnostic): param0 is
// 'int __attribute__((vector_size(16))) addrspace(1)*', param1 addrspace(3).
typedef int v4i_ls __attribute__((vector_size(16)));
typedef __attribute__((address_space(1))) v4i_ls* g_v4i_p;
typedef __attribute__((address_space(3))) v4i_ls* l_v4i_p;

// Direct global -> LDS async copy, 16 bytes per lane, tracked by ASYNCcnt.
__device__ __forceinline__ void async_g2l_b128(const void* g, void* l) {
    __builtin_amdgcn_global_load_async_to_lds_b128(
        (g_v4i_p)g, (l_v4i_p)l, 0, 0);
}
#endif

// ---------------------------------------------------------------------------
// fp32 -> bf16 weight conversion (one thread per float4 chunk)
// ---------------------------------------------------------------------------
__global__ __launch_bounds__(256) void cvt_bf16_kernel(
    const float* __restrict__ in, unsigned short* __restrict__ out, long long n4)
{
    long long i = (long long)blockIdx.x * blockDim.x + threadIdx.x;
    if (i >= n4) return;
    float4 v = ((const float4*)in)[i];
    ushort4 o;
    o.x = f2bf(v.x); o.y = f2bf(v.y); o.z = f2bf(v.z); o.w = f2bf(v.w);
    ((ushort4*)out)[i] = o;
}

// ---------------------------------------------------------------------------
// LayerNorm over H=1024, one block (256 threads) per token row, bf16 out.
// ---------------------------------------------------------------------------
__global__ __launch_bounds__(256) void ln_kernel(
    const float* __restrict__ x, const float* __restrict__ w,
    const float* __restrict__ b, unsigned short* __restrict__ out)
{
    const int row = blockIdx.x;
    const int t   = threadIdx.x;
    const float4 v = ((const float4*)(x + (size_t)row * 1024))[t];
    float s  = v.x + v.y + v.z + v.w;
    float s2 = v.x * v.x + v.y * v.y + v.z * v.z + v.w * v.w;
    for (int m = 16; m >= 1; m >>= 1) {
        s  += __shfl_xor(s,  m, 32);
        s2 += __shfl_xor(s2, m, 32);
    }
    __shared__ float red[16];
    if ((t & 31) == 0) { red[t >> 5] = s; red[8 + (t >> 5)] = s2; }
    __syncthreads();
    float ts = 0.f, ts2 = 0.f;
    for (int i = 0; i < 8; i++) { ts += red[i]; ts2 += red[8 + i]; }
    const float mu  = ts * (1.0f / 1024.0f);
    const float inv = rsqrtf(ts2 * (1.0f / 1024.0f) - mu * mu + 1e-5f);
    const float4 wv = ((const float4*)w)[t];
    const float4 bv = ((const float4*)b)[t];
    ushort4 o;
    o.x = f2bf((v.x - mu) * inv * wv.x + bv.x);
    o.y = f2bf((v.y - mu) * inv * wv.y + bv.y);
    o.z = f2bf((v.z - mu) * inv * wv.z + bv.z);
    o.w = f2bf((v.w - mu) * inv * wv.w + bv.w);
    ((ushort4*)(out + (size_t)row * 1024))[t] = o;
}

// ---------------------------------------------------------------------------
// Tiled bf16 WMMA GEMM:  out[M,N] = A[M,K] * W[N,K]^T + bias (+gelu) (+res)
// 128x128 workgroup tile, K-step 32, LDS-staged, 8 waves each 32(M)x64(N).
// ---------------------------------------------------------------------------
__global__ __launch_bounds__(256) void gemm_bf16_kernel(
    const unsigned short* __restrict__ A,     // [M,K] bf16
    const unsigned short* __restrict__ W,     // [N,K] bf16
    const float* __restrict__ bias,           // [N]
    const float* __restrict__ res,            // [M,N] fp32 or nullptr
    void* __restrict__ out,                   // fp32 or bf16 per out_bf16
    int M, int N, int K, int act, int out_bf16)
{
    constexpr int LDT = 40;                   // padded LDS row stride (elems)
    __shared__ unsigned short ldsA[128 * LDT];
    __shared__ unsigned short ldsB[128 * LDT];

    const int n0   = blockIdx.x * 128;
    const int m0   = blockIdx.y * 128;
    const int t    = threadIdx.x;
    const int w    = t >> 5;
    const int lane = t & 31;
    const int half = lane >> 4;
    const int ln16 = lane & 15;
    const int wm   = w & 3;                   // wave M slot: 0..3 (x32 rows)
    const int wn   = w >> 2;                  // wave N slot: 0..1 (x64 cols)
    const int sr   = t >> 1;                  // staging row 0..127
    const int sc   = (t & 1) << 4;            // staging col 0 or 16

    const v8f vz = {0.f,0.f,0.f,0.f,0.f,0.f,0.f,0.f};
    v8f acc[2][4];
    for (int i = 0; i < 2; i++)
        for (int j = 0; j < 4; j++) acc[i][j] = vz;

    for (int k0 = 0; k0 < K; k0 += 32) {
        const unsigned short* gA = A + (size_t)(m0 + sr) * K + k0 + sc;
        const unsigned short* gW = W + (size_t)(n0 + sr) * K + k0 + sc;
#if USE_ASYNC_LDS
        __syncthreads();                      // previous iteration's reads done
        async_g2l_b128(gA,     ldsA + sr * LDT + sc);
        async_g2l_b128(gA + 8, ldsA + sr * LDT + sc + 8);
        async_g2l_b128(gW,     ldsB + sr * LDT + sc);
        async_g2l_b128(gW + 8, ldsB + sr * LDT + sc + 8);
        __builtin_amdgcn_s_wait_asynccnt(0);
        __syncthreads();                      // tile visible to all waves
#else
        const uint4* ga = (const uint4*)gA;
        const uint4* gb = (const uint4*)gW;
        uint4 a0 = ga[0], a1 = ga[1];
        uint4 b0 = gb[0], b1 = gb[1];
        if (k0 + 32 < K) {                    // hint next K tile into cache
            __builtin_prefetch(gA + 32, 0, 1);
            __builtin_prefetch(gW + 32, 0, 1);
        }
        __syncthreads();
        *(uint4*)(ldsA + sr * LDT + sc)     = a0;
        *(uint4*)(ldsA + sr * LDT + sc + 8) = a1;
        *(uint4*)(ldsB + sr * LDT + sc)     = b0;
        *(uint4*)(ldsB + sr * LDT + sc + 8) = b1;
        __syncthreads();
#endif

        Frag fa[2], fb[4];
        for (int i = 0; i < 2; i++) {
            // A operand (16x32): lane m=ln16, K chunks half*8 and 16+half*8
            const uint4* p = (const uint4*)(ldsA + (wm * 32 + i * 16 + ln16) * LDT);
            fa[i].q[0] = p[half];
            fa[i].q[1] = p[2 + half];
        }
        for (int j = 0; j < 4; j++) {
            // B operand (32x16): lane n=ln16, K chunks half*8 and 16+half*8
            const uint4* p = (const uint4*)(ldsB + (wn * 64 + j * 16 + ln16) * LDT);
            fb[j].q[0] = p[half];
            fb[j].q[1] = p[2 + half];
        }
        for (int i = 0; i < 2; i++)
            for (int j = 0; j < 4; j++)
                acc[i][j] = __builtin_amdgcn_wmma_f32_16x16x32_bf16(
                    false, fa[i].v, false, fb[j].v, (short)0, acc[i][j],
                    false, false);
    }

    // Epilogue: D layout -> element (vgpr v, lane): row = v + 8*half, col = ln16
    for (int i = 0; i < 2; i++) {
        const int mrow = m0 + wm * 32 + i * 16 + half * 8;
        for (int j = 0; j < 4; j++) {
            const int ncol = n0 + wn * 64 + j * 16 + ln16;
            const float bcol = bias[ncol];
            for (int v = 0; v < 8; v++) {
                float val = acc[i][j][v] + bcol;
                if (act) val = gelu_exact(val);
                const size_t idx = (size_t)(mrow + v) * N + ncol;
                if (res) val += res[idx];
                if (out_bf16) ((unsigned short*)out)[idx] = f2bf(val);
                else          ((float*)out)[idx] = val;
            }
        }
    }
}

// ---------------------------------------------------------------------------
// Flash-style causal attention with ALiBi, bf16 WMMA.
// qkv: [B*S, 3*NH*DH] bf16 (q | k | v blocks, head-major within each).
// One block per (128 q rows, head, batch); each wave owns 16 q rows.
// Effective score: s = qk/sqrt(DH) + alibi[kv], masked to -10000, softmax.
// ---------------------------------------------------------------------------
__global__ __launch_bounds__(256) void attn_kernel(
    const unsigned short* __restrict__ qkv,
    const float* __restrict__ alibi,          // [B, NH, S]
    unsigned short* __restrict__ ctx)         // [B*S, NH*DH] bf16
{
    constexpr int S = 2048, NH = 16;
    __shared__ unsigned short ldsVT[64][72];        // V^T tile: [d][kv]
    __shared__ unsigned short ldsP[8][16][72];      // per-wave P: [q][kv]

    const int qblk = blockIdx.x;
    const int h    = blockIdx.y;
    const int b    = blockIdx.z;
    const int t    = threadIdx.x;
    const int w    = t >> 5;
    const int lane = t & 31;
    const int half = lane >> 4;
    const int ln16 = lane & 15;
    const int qbase = qblk * 128 + w * 16;

    // Preload Q fragments (A operand, 16 q rows x DH=64 in two K-steps of 32)
    Frag fq[2];
    {
        const unsigned short* qp =
            qkv + (size_t)(b * S + qbase + ln16) * 3072 + h * 64;
        for (int ks = 0; ks < 2; ks++) {
            const uint4* p = (const uint4*)(qp + ks * 32);
            fq[ks].q[0] = p[half];
            fq[ks].q[1] = p[2 + half];
        }
    }

    const v8f vz = {0.f,0.f,0.f,0.f,0.f,0.f,0.f,0.f};
    v8f o[4];
    for (int i = 0; i < 4; i++) o[i] = vz;
    float rm[8], rl[8];
    for (int v = 0; v < 8; v++) { rm[v] = -1e30f; rl[v] = 0.f; }

    const int nkv = qblk * 2 + 2;                  // kv blocks of 64 up to diag
    for (int kvb = 0; kvb < nkv; kvb++) {
        const int kv0 = kvb * 64;

        // ---- cooperative stage of V^T into LDS -------------------------
        __syncthreads();
        {
            const int r  = t >> 2;                 // kv row 0..63
            const int dc = (t & 3) << 4;           // d chunk 0/16/32/48
            const uint4* vp = (const uint4*)(
                qkv + (size_t)(b * S + kv0 + r) * 3072 + 2048 + h * 64 + dc);
            Frag tv; tv.q[0] = vp[0]; tv.q[1] = vp[1];
            for (int jj = 0; jj < 16; jj++) ldsVT[dc + jj][r] = tv.s[jj];
        }
        __syncthreads();

        // ---- S = Q K^T (16 x 64), K fragments straight from global ----
        v8f sf[4];
        for (int nf = 0; nf < 4; nf++) sf[nf] = vz;
        for (int ks = 0; ks < 2; ks++) {
            for (int nf = 0; nf < 4; nf++) {
                Frag fk;
                const uint4* p = (const uint4*)(
                    qkv + (size_t)(b * S + kv0 + nf * 16 + ln16) * 3072 +
                    1024 + h * 64 + ks * 32);
                fk.q[0] = p[half];
                fk.q[1] = p[2 + half];
                sf[nf] = __builtin_amdgcn_wmma_f32_16x16x32_bf16(
                    false, fq[ks].v, false, fk.v, (short)0, sf[nf],
                    false, false);
            }
        }

        // ---- scale + alibi + causal mask, running softmax -------------
        float tmax[8];
        for (int v = 0; v < 8; v++) tmax[v] = -1e30f;
        for (int nf = 0; nf < 4; nf++) {
            const int kg = kv0 + nf * 16 + ln16;
            const float al = alibi[((size_t)b * NH + h) * S + kg];
            for (int v = 0; v < 8; v++) {
                float sv = sf[nf][v] * 0.125f + al;
                const int qg = qbase + half * 8 + v;
                if (kg > qg) sv = -10000.0f;
                sf[nf][v] = sv;
                tmax[v] = fmaxf(tmax[v], sv);
            }
        }
        for (int v = 0; v < 8; v++) {
            for (int m = 1; m < 16; m <<= 1)
                tmax[v] = fmaxf(tmax[v], __shfl_xor(tmax[v], m, 32));
            const float mnew = fmaxf(rm[v], tmax[v]);
            const float cs = __expf(rm[v] - mnew);
            rm[v] = mnew;
            rl[v] *= cs;
            for (int nf2 = 0; nf2 < 4; nf2++) o[nf2][v] *= cs;
        }
        float psum[8];
        for (int v = 0; v < 8; v++) psum[v] = 0.f;
        for (int nf = 0; nf < 4; nf++)
            for (int v = 0; v < 8; v++) {
                const float p = __expf(sf[nf][v] - rm[v]);
                sf[nf][v] = p;
                psum[v] += p;
            }
        for (int v = 0; v < 8; v++) {
            float pv = psum[v];
            for (int m = 1; m < 16; m <<= 1) pv += __shfl_xor(pv, m, 32);
            rl[v] += pv;
        }

        // ---- P (D layout) -> LDS -> A layout, then O += P V ----------
        for (int nf = 0; nf < 4; nf++)
            for (int v = 0; v < 8; v++)
                ldsP[w][half * 8 + v][nf * 16 + ln16] = f2bf(sf[nf][v]);

        for (int ks = 0; ks < 2; ks++) {
            Frag fp;
            const uint4* pp = (const uint4*)(&ldsP[w][ln16][0]);
            fp.q[0] = pp[ks * 4 + half];
            fp.q[1] = pp[ks * 4 + 2 + half];
            for (int nf2 = 0; nf2 < 4; nf2++) {
                Frag fv;
                const uint4* vp2 = (const uint4*)(&ldsVT[nf2 * 16 + ln16][0]);
                fv.q[0] = vp2[ks * 4 + half];
                fv.q[1] = vp2[ks * 4 + 2 + half];
                o[nf2] = __builtin_amdgcn_wmma_f32_16x16x32_bf16(
                    false, fp.v, false, fv.v, (short)0, o[nf2], false, false);
            }
        }
    }

    // ---- normalize and write ctx (bf16) -------------------------------
    for (int nf2 = 0; nf2 < 4; nf2++) {
        for (int v = 0; v < 8; v++) {
            const float val = o[nf2][v] / rl[v];
            ctx[(size_t)(b * S + qbase + half * 8 + v) * 1024 +
                h * 64 + nf2 * 16 + ln16] = f2bf(val);
        }
    }
}

// ---------------------------------------------------------------------------
// Host orchestration
// ---------------------------------------------------------------------------
extern "C" void kernel_launch(void* const* d_in, const int* in_sizes, int n_in,
                              void* d_out, int out_size, void* d_ws, size_t ws_size,
                              hipStream_t stream)
{
    (void)in_sizes; (void)n_in; (void)out_size; (void)ws_size;
    const float* x_in    = (const float*)d_in[0];
    const float* alibi   = (const float*)d_in[1];
    const float* qkv_w   = (const float*)d_in[2];
    const float* qkv_b   = (const float*)d_in[3];
    const float* dense_w = (const float*)d_in[4];
    const float* dense_b = (const float*)d_in[5];
    const float* mlp_w1  = (const float*)d_in[6];
    const float* mlp_b1  = (const float*)d_in[7];
    const float* mlp_w2  = (const float*)d_in[8];
    const float* mlp_b2  = (const float*)d_in[9];
    const float* ln1_w   = (const float*)d_in[10];
    const float* ln1_b   = (const float*)d_in[11];
    const float* ln2_w   = (const float*)d_in[12];
    const float* ln2_b   = (const float*)d_in[13];

    constexpr int L = 2, B = 2, S = 2048, H = 1024, FF = 4096, QKV = 3072;
    constexpr size_t M = (size_t)B * S;  // 4096 token rows

    char* ws = (char*)d_ws;
    size_t off = 0;
    auto alloc = [&](size_t bytes) -> void* {
        void* p = ws + off;
        off += (bytes + 255) & ~(size_t)255;
        return p;
    };
    unsigned short* wqkv = (unsigned short*)alloc((size_t)L * QKV * H * 2);
    unsigned short* wdns = (unsigned short*)alloc((size_t)L * H * H * 2);
    unsigned short* w1   = (unsigned short*)alloc((size_t)L * FF * H * 2);
    unsigned short* w2   = (unsigned short*)alloc((size_t)L * H * FF * 2);
    unsigned short* hbuf = (unsigned short*)alloc(M * H * 2);
    unsigned short* qkvb = (unsigned short*)alloc(M * QKV * 2);
    unsigned short* ctxb = (unsigned short*)alloc(M * H * 2);
    unsigned short* mbuf = (unsigned short*)alloc(M * FF * 2);
    float* xa = (float*)alloc(M * H * 4);
    float* xb = (float*)alloc(M * H * 4);

    auto cvt = [&](const float* in, unsigned short* out, long long n) {
        long long n4 = n / 4;
        cvt_bf16_kernel<<<dim3((unsigned)((n4 + 255) / 256)), dim3(256), 0,
                          stream>>>(in, out, n4);
    };
    cvt(qkv_w,   wqkv, (long long)L * QKV * H);
    cvt(dense_w, wdns, (long long)L * H * H);
    cvt(mlp_w1,  w1,   (long long)L * FF * H);
    cvt(mlp_w2,  w2,   (long long)L * H * FF);

    auto gemm = [&](const unsigned short* A, const unsigned short* Wm,
                    const float* bias, const float* res, void* out,
                    int N, int K, int act, int obf) {
        gemm_bf16_kernel<<<dim3(N / 128, (int)(M / 128)), dim3(256), 0,
                           stream>>>(A, Wm, bias, res, out, (int)M, N, K, act,
                                     obf);
    };

    const float* xcur = x_in;
    for (int l = 0; l < L; l++) {
        ln_kernel<<<dim3((int)M), dim3(256), 0, stream>>>(
            xcur, ln1_w + (size_t)l * H, ln1_b + (size_t)l * H, hbuf);
        gemm(hbuf, wqkv + (size_t)l * QKV * H, qkv_b + (size_t)l * QKV,
             nullptr, qkvb, QKV, H, 0, 1);
        attn_kernel<<<dim3(S / 128, 16, B), dim3(256), 0, stream>>>(
            qkvb, alibi, ctxb);
        gemm(ctxb, wdns + (size_t)l * H * H, dense_b + (size_t)l * H,
             xcur, xa, H, H, 0, 0);
        ln_kernel<<<dim3((int)M), dim3(256), 0, stream>>>(
            xa, ln2_w + (size_t)l * H, ln2_b + (size_t)l * H, hbuf);
        gemm(hbuf, w1 + (size_t)l * FF * H, mlp_b1 + (size_t)l * FF,
             nullptr, mbuf, FF, H, 1, 1);
        float* xnext = (l == L - 1) ? (float*)d_out : xb;
        gemm(mbuf, w2 + (size_t)l * H * FF, mlp_b2 + (size_t)l * H,
             xa, xnext, H, FF, 0, 0);
        xcur = xnext;
    }
}
